// OutputTransition_8804682957054
// MI455X (gfx1250) — matrix operand order repaired and checked by reading, not compiled, not run on hardware
//
#include <hip/hip_runtime.h>
#include <hip/hip_bf16.h>

// ---------------- problem constants ----------------
#define N_VOX   500000
#define CIN     32
#define COUT    32
#define NTAP    125
#define EPS_BN  1e-5f

// ---------------- conv kernel config ----------------
#define TILE_M          16
#define NTILES          ((N_VOX + TILE_M - 1) / TILE_M)   // 31250
#define TPG             4                                  // tiles per wave group
#define NGROUPS         ((NTILES + TPG - 1) / TPG)         // 7813 (last group partial)
#define CONV_BLOCKS     250
#define CONV_THREADS    1024
#define WAVES_PER_BLOCK (CONV_THREADS / 32)

#define BPACK_ELEMS (NTAP * 2 * 32 * 16)    // 128000 bf16 elements
#define BPACK_BYTES (BPACK_ELEMS * 2)       // 256000 B
#define RED_BYTES   (CONV_THREADS * 4 * 4)  // 16384 B
#define SMEM_BYTES  (BPACK_BYTES + RED_BYTES)

// ---------------- workspace layout (bytes) ----------------
#define OFF_XH    0ull          // N*32 bf16          = 32,000,000 B
#define OFF_CONV  32000000ull   // N*32 f32           = 64,000,000 B
#define OFF_BP    96000000ull   // B-pack bf16        =    256,000 B
#define OFF_PART  96256000ull   // 250*64 f32         =     64,000 B
#define OFF_STATS 96320000ull   // mean[32], rstd[32] =        256 B

typedef __bf16 bf16_t;
typedef __attribute__((ext_vector_type(16))) __bf16 v16bf;
typedef __attribute__((ext_vector_type(8)))  __bf16 v8bf;
typedef __attribute__((ext_vector_type(8)))  float  v8f;

union Frag16 { v16bf v; v8bf h[2]; };

// ============================================================
// Kernel 1: cast x -> bf16 rows; pack W_conv -> WMMA B-fragment
// B-pack linear index: t*1024 + h*512 + L*16 + e
//   lane L, elem e -> K = e + 16*(L>>4), N = (L&15) + 16*h
// ============================================================
__global__ __launch_bounds__(256)
void sp3d_prep(const float* __restrict__ x,
               const float* __restrict__ Wc,
               bf16_t* __restrict__ xh,
               bf16_t* __restrict__ bpack)
{
    const long long i  = (long long)blockIdx.x * blockDim.x + threadIdx.x;
    const long long NX = (long long)N_VOX * CIN;
    if (i < NX) { xh[i] = (bf16_t)x[i]; return; }
    const long long j = i - NX;
    if (j < (long long)BPACK_ELEMS) {
        const int e  = (int)(j & 15);
        const int L  = (int)((j >> 4) & 31);
        const int h  = (int)((j >> 9) & 1);
        const int t  = (int)(j >> 10);
        const int sL = L >> 4;
        const int k  = e + 16 * sL;
        const int n  = (L & 15) + 16 * h;
        bpack[j] = (bf16_t)Wc[(t * CIN + k) * COUT + n];
    }
}

// ============================================================
// Group body for the conv kernel.
// GUARD=false: all 64 voxels of the group are in range (no bounds VALU
// in the tap loop). GUARD=true: last partial group only.
// ============================================================
template <bool GUARD>
__device__ __forceinline__ void conv_group(
    const int base, const int s, const int col, const int lane,
    const int*    __restrict__ nbr,
    const bf16_t* __restrict__ xh,
    const char*   smem,
    const float bias0, const float bias1,
    float* __restrict__ conv_out,
    float& ls0, float& lq0, float& ls1, float& lq1)
{
    v8bf zero8;
    #pragma unroll
    for (int i = 0; i < 8; ++i) zero8[i] = (bf16_t)0.0f;

    v8f acc[TPG][2];
    #pragma unroll
    for (int tt = 0; tt < TPG; ++tt)
        #pragma unroll
        for (int r = 0; r < 8; ++r) { acc[tt][0][r] = bias0; acc[tt][1][r] = bias1; }

    // Per-tile neighbor-row pointers (clamped so GUARDed lanes never
    // form an out-of-range address; they are masked off anyway).
    const int* nb[TPG];
    bool mok[TPG];
    #pragma unroll
    for (int tt = 0; tt < TPG; ++tt) {
        const int m = base + tt * TILE_M + col;
        mok[tt] = !GUARD || (m < N_VOX);
        const int mc = GUARD ? (m < N_VOX ? m : 0) : m;
        nb[tt] = nbr + (size_t)mc * NTAP;
    }

    const char* bptr = smem + lane * 32;   // running B-fragment pointer

    #pragma unroll 1
    for (int t = 0; t < NTAP; ++t) {
        // B fragments for this tap (both COUT halves) from LDS.
        Frag16 b0, b1;
        b0.h[0] = ((const v8bf*)bptr)[0];
        b0.h[1] = ((const v8bf*)bptr)[1];
        b1.h[0] = ((const v8bf*)(bptr + 1024))[0];
        b1.h[1] = ((const v8bf*)(bptr + 1024))[1];
        bptr += 2048;

        // Gather all A fragments first (batches the global loads).
        Frag16 a[TPG];
        #pragma unroll
        for (int tt = 0; tt < TPG; ++tt) {
            int idx = -1;
            if (mok[tt]) idx = nb[tt][t];
            if (idx >= 0) {
                const bf16_t* row = xh + (size_t)idx * CIN;     // 64 B bf16 row
                a[tt].h[0] = *(const v8bf*)(row + 16 * s);      // K = 16s .. 16s+7
                a[tt].h[1] = *(const v8bf*)(row + 16 * s + 8);  // K = 16s+8 .. 16s+15
            } else {
                a[tt].h[0] = zero8; a[tt].h[1] = zero8;
            }
        }

        // WMMA burst: 8 independent 16x16x32 bf16 MACs.
        #pragma unroll
        for (int tt = 0; tt < TPG; ++tt) {
            acc[tt][0] = __builtin_amdgcn_wmma_f32_16x16x32_bf16(
                false, a[tt].v, false, b0.v, (short)0, acc[tt][0], false, false);
            acc[tt][1] = __builtin_amdgcn_wmma_f32_16x16x32_bf16(
                false, a[tt].v, false, b1.v, (short)0, acc[tt][1], false, false);
        }
    }

    // Write conv output and accumulate per-lane channel stats.
    // C/D layout: vgpr r, lane half s -> M = r + 8s; N = col (+16 for acc1)
    #pragma unroll
    for (int tt = 0; tt < TPG; ++tt) {
        #pragma unroll
        for (int r = 0; r < 8; ++r) {
            const int m = base + tt * TILE_M + r + 8 * s;
            if (!GUARD || m < N_VOX) {
                const float v0 = acc[tt][0][r];
                const float v1 = acc[tt][1][r];
                conv_out[(size_t)m * COUT + col]      = v0;
                conv_out[(size_t)m * COUT + 16 + col] = v1;
                ls0 += v0; lq0 += v0 * v0;
                ls1 += v1; lq1 += v1 * v1;
            }
        }
    }
}

// ============================================================
// Kernel 2: gather + per-tap WMMA GEMM, bias init, partial stats
// ============================================================
__global__ __launch_bounds__(CONV_THREADS)
void sp3d_conv_wmma(const int*    __restrict__ nbr,
                    const float*  __restrict__ b_conv,
                    const bf16_t* __restrict__ xh,
                    const bf16_t* __restrict__ bpack,
                    float*        __restrict__ conv_out,
                    float*        __restrict__ partials)
{
    extern __shared__ char smem[];
    const int tid = threadIdx.x;

    // Stage 256 KB of packed B-matrices into LDS (b128 copies).
    {
        const uint4* src = (const uint4*)bpack;
        uint4*       dst = (uint4*)smem;
        for (int i = tid; i < BPACK_BYTES / 16; i += CONV_THREADS) dst[i] = src[i];
    }
    __syncthreads();

    const int lane = tid & 31;
    const int wave = tid >> 5;
    const int s    = lane >> 4;   // lane half selects K sub-range
    const int col  = lane & 15;   // M-row for A gather / N-column for C

    const float bias0 = b_conv[col];
    const float bias1 = b_conv[16 + col];

    float ls0 = 0.f, lq0 = 0.f, ls1 = 0.f, lq1 = 0.f;

    const int gwave  = blockIdx.x * WAVES_PER_BLOCK + wave;
    const int nwaves = CONV_BLOCKS * WAVES_PER_BLOCK;

    for (int grp = gwave; grp < NGROUPS; grp += nwaves) {
        const int base = grp * (TPG * TILE_M);
        if (grp == NGROUPS - 1) {
            conv_group<true>(base, s, col, lane, nbr, xh, smem,
                             bias0, bias1, conv_out, ls0, lq0, ls1, lq1);
        } else {
            conv_group<false>(base, s, col, lane, nbr, xh, smem,
                              bias0, bias1, conv_out, ls0, lq0, ls1, lq1);
        }
    }

    // Block-level ordered reduction of (sum, sumsq) per channel.
    float* red = (float*)(smem + BPACK_BYTES);
    red[tid * 4 + 0] = ls0; red[tid * 4 + 1] = lq0;
    red[tid * 4 + 2] = ls1; red[tid * 4 + 3] = lq1;
    __syncthreads();
    if (tid < COUT) {
        const int c    = tid;
        const int l0   = (c < 16) ? c : (c - 16);
        const int slot = (c < 16) ? 0 : 2;
        float S = 0.f, Q = 0.f;
        for (int w = 0; w < WAVES_PER_BLOCK; ++w) {
            const float* r0 = red + ((w * 32 + l0) * 4);
            const float* r1 = red + ((w * 32 + l0 + 16) * 4);
            S += r0[slot] + r1[slot];
            Q += r0[slot + 1] + r1[slot + 1];
        }
        partials[blockIdx.x * 64 + c]      = S;
        partials[blockIdx.x * 64 + 32 + c] = Q;
    }
}

// ============================================================
// Kernel 3: reduce per-block partials -> mean, rstd
// ============================================================
__global__ __launch_bounds__(64)
void sp3d_stats(const float* __restrict__ partials, float* __restrict__ stats)
{
    const int c = threadIdx.x;
    if (c >= COUT) return;
    float S = 0.f, Q = 0.f;
    for (int b = 0; b < CONV_BLOCKS; ++b) {
        S += partials[b * 64 + c];
        Q += partials[b * 64 + 32 + c];
    }
    const float inv  = 1.0f / (float)N_VOX;
    const float mean = S * inv;
    const float var  = fmaxf(Q * inv - mean * mean, 0.f);
    stats[c]      = mean;
    stats[32 + c] = rsqrtf(var + EPS_BN);
}

// ============================================================
// Kernel 4: BN + ELU + 32x32 linear (W_lin staged in LDS)
// ============================================================
__global__ __launch_bounds__(256)
void sp3d_finalize(const float* __restrict__ conv_out,
                   const float* __restrict__ stats,
                   const float* __restrict__ gamma,
                   const float* __restrict__ beta,
                   const float* __restrict__ W_lin,
                   const float* __restrict__ b_lin,
                   float* __restrict__ out)
{
    __shared__ float sW[COUT * COUT];
    __shared__ float sScale[COUT], sShift[COUT], sBl[COUT];
    const int tid = threadIdx.x;
    for (int i = tid; i < COUT * COUT; i += 256) sW[i] = W_lin[i];
    if (tid < COUT) {
        const float mean = stats[tid];
        const float sc   = stats[32 + tid] * gamma[tid];
        sScale[tid] = sc;
        sShift[tid] = beta[tid] - mean * sc;
        sBl[tid]    = b_lin[tid];
    }
    __syncthreads();

    const int v = blockIdx.x * 256 + tid;
    if (v >= N_VOX) return;

    float e[COUT];
    #pragma unroll
    for (int c = 0; c < COUT; ++c) {
        const float y = conv_out[(size_t)v * COUT + c] * sScale[c] + sShift[c];
        e[c] = (y > 0.f) ? y : expm1f(y);          // ELU, alpha = 1
    }
    #pragma unroll 4
    for (int o = 0; o < COUT; ++o) {
        float a = sBl[o];
        #pragma unroll
        for (int c = 0; c < COUT; ++c) a += e[c] * sW[o * COUT + c];
        out[(size_t)v * COUT + o] = a;             // out = e @ W_lin.T + b_lin
    }
}

// ============================================================
extern "C" void kernel_launch(void* const* d_in, const int* in_sizes, int n_in,
                              void* d_out, int out_size, void* d_ws, size_t ws_size,
                              hipStream_t stream)
{
    (void)in_sizes; (void)n_in; (void)out_size; (void)ws_size;

    const float* x     = (const float*)d_in[0];
    const int*   nbr   = (const int*)  d_in[1];
    const float* Wc    = (const float*)d_in[2];
    const float* bc    = (const float*)d_in[3];
    const float* gamma = (const float*)d_in[4];
    const float* beta  = (const float*)d_in[5];
    const float* Wl    = (const float*)d_in[6];
    const float* bl    = (const float*)d_in[7];
    float*       out   = (float*)d_out;

    char* ws = (char*)d_ws;
    bf16_t* xh    = (bf16_t*)(ws + OFF_XH);
    float*  conv  = (float*) (ws + OFF_CONV);
    bf16_t* bpack = (bf16_t*)(ws + OFF_BP);
    float*  part  = (float*) (ws + OFF_PART);
    float*  stats = (float*) (ws + OFF_STATS);

    // Allow > default dynamic LDS (256 KB B-pack + 16 KB reduction scratch).
    hipFuncSetAttribute((const void*)sp3d_conv_wmma,
                        hipFuncAttributeMaxDynamicSharedMemorySize, SMEM_BYTES);

    const long long prep_total  = (long long)N_VOX * CIN + BPACK_ELEMS;
    const int       prep_blocks = (int)((prep_total + 255) / 256);

    sp3d_prep<<<prep_blocks, 256, 0, stream>>>(x, Wc, xh, bpack);
    sp3d_conv_wmma<<<CONV_BLOCKS, CONV_THREADS, SMEM_BYTES, stream>>>(
        nbr, bc, xh, bpack, conv, part);
    sp3d_stats<<<1, 64, 0, stream>>>(part, stats);
    sp3d_finalize<<<(N_VOX + 255) / 256, 256, 0, stream>>>(
        conv, stats, gamma, beta, Wl, bl, out);
}
